// FPN_ESGRU_K_L_U_19000935318218
// MI455X (gfx1250) — matrix operand order repaired and checked by reading, not compiled, not run on hardware
//
#include <hip/hip_runtime.h>
#include <hip/hip_bf16.h>
#include <math.h>

// ---------------------------------------------------------------------------
// FPN + grouped-GRU + attention-pool network for MI455X (gfx1250, wave32).
// GEMM-shaped stages (3 encoder convs, 3 FPN pointwise convs, attention
// projection) run on v_wmma_f32_16x16x32_f16 with f32 accumulation and fused
// bias/BN/GELU epilogues.  Depthwise convs / upsample-adds are elementwise.
// The sequential GRU runs one wave per batch element with h in registers.
//
// Input flattening assumption (setup_inputs dict order): x first, then params
// in insertion order (see index table in kernel_launch).
// Workspace requirement: ~78 MiB (buffers are reused along the dataflow).
// ---------------------------------------------------------------------------

typedef __attribute__((ext_vector_type(16))) _Float16 v16h;
typedef __attribute__((ext_vector_type(8)))  float    v8f;
typedef _Float16 f16;

#define DEVI __device__ __forceinline__

static constexpr int BATCH = 128;

union AFrag { v16h v; uint4 q[2]; };

DEVI float gelu_exact(float v) {
    return 0.5f * v * (1.0f + erff(v * 0.70710678118654752f));
}

// ---------------------------------------------------------------------------
// Implicit-GEMM conv (also used for 1x1 pointwise).  Data layout: channel-last
// f16 (B, L, C).  K ordering: k = t*CIN + ci, zero-padded to NCHUNK*32.
// One wave per 16-row M tile, all COUT/16 N tiles accumulated in registers.
// ---------------------------------------------------------------------------
template <int CIN, int COUT, int KSZ, int STRIDE, int PAD, int LIN, int LOUT, int NCHUNK>
__global__ __launch_bounds__(256) void conv_wmma(
    const f16* __restrict__ inh,   // (B, LIN, CIN)
    const f16* __restrict__ wp,    // (COUT, NCHUNK*32) f16, K-packed
    const float* __restrict__ bnA, // per-channel scale   (BN*conv-bias fused)
    const float* __restrict__ bnB, // per-channel shift
    f16* __restrict__ outh)        // (B, LOUT, COUT)
{
    constexpr int KPAD = NCHUNK * 32;
    constexpr int KTOT = KSZ * CIN;
    constexpr int NT   = COUT / 16;

    const int lane = threadIdx.x & 31;
    const int wave = threadIdx.x >> 5;
    const int mtile = blockIdx.x * 8 + wave;
    if (mtile >= (BATCH * LOUT) / 16) return;

    const int half = lane >> 4;
    const int n    = lane & 15;            // A-row within tile == D-column
    const int grow0 = mtile * 16;
    const int b   = grow0 / LOUT;
    const int lo0 = grow0 % LOUT;
    const int lo  = lo0 + n;               // output position of this A-row

    const f16* inb = inh + (size_t)b * LIN * CIN;

    v8f acc[NT];
#pragma unroll
    for (int i = 0; i < NT; ++i) acc[i] = (v8f){0.f,0.f,0.f,0.f,0.f,0.f,0.f,0.f};

    for (int c = 0; c < NCHUNK; ++c) {
        AFrag a;
#pragma unroll
        for (int blk = 0; blk < 2; ++blk) {
            const int k8 = 32 * c + 16 * half + 8 * blk;   // 8-aligned K run
            uint4 q = {0u, 0u, 0u, 0u};
            if (k8 < KTOT) {
                const int t  = k8 / CIN;
                const int ci = k8 % CIN;
                const int li = STRIDE * lo + t - PAD;
                if (li >= 0 && li < LIN)
                    q = *(const uint4*)(inb + (size_t)li * CIN + ci);
            }
            a.q[blk] = q;
        }
#pragma unroll
        for (int nt = 0; nt < NT; ++nt) {
            const int co = nt * 16 + n;
            v16h bf = *(const v16h*)(wp + (size_t)co * KPAD + 32 * c + 16 * half);
            acc[nt] = __builtin_amdgcn_wmma_f32_16x16x32_f16(
                false, a.v, false, bf, (short)0, acc[nt], false, false);
        }
    }

    f16* outb = outh + (size_t)b * LOUT * COUT;
#pragma unroll
    for (int nt = 0; nt < NT; ++nt) {
        const int co = nt * 16 + n;
        const float A = bnA[co];
        const float B = bnB[co];
#pragma unroll
        for (int r = 0; r < 8; ++r) {
            const int m = r + 8 * half;
            float v = acc[nt][r] * A + B;
            v = gelu_exact(v);
            outb[(size_t)(lo0 + m) * COUT + co] = (f16)v;
        }
    }
}

// ---------------------------------------------------------------------------
// Attention score:  s = tanh(g @ W1^T + b1) @ W2^T + b2  — WMMA on the 32x32
// projection, dot with w2 + lane-shuffle reduction fused in the epilogue.
// g: (B, 2048, 32) f16.  s: (B, 2048) f32.
// ---------------------------------------------------------------------------
__global__ __launch_bounds__(256) void attn_score(
    const f16* __restrict__ gh, const f16* __restrict__ w1p,
    const float* __restrict__ b1, const float* __restrict__ w2,
    const float* __restrict__ b2, float* __restrict__ s)
{
    const int lane = threadIdx.x & 31;
    const int wave = threadIdx.x >> 5;
    const int mtile = blockIdx.x * 8 + wave;
    if (mtile >= (BATCH * 2048) / 16) return;

    const int half = lane >> 4;
    const int n    = lane & 15;
    const int grow0 = mtile * 16;
    const int b   = grow0 / 2048;
    const int lo0 = grow0 % 2048;
    const f16* inb = gh + (size_t)b * 2048 * 32;

    AFrag a;
#pragma unroll
    for (int blk = 0; blk < 2; ++blk) {
        const int k8 = 16 * half + 8 * blk;
        a.q[blk] = *(const uint4*)(inb + (size_t)(lo0 + n) * 32 + k8);
    }

    v8f acc[2];
#pragma unroll
    for (int i = 0; i < 2; ++i) acc[i] = (v8f){0.f,0.f,0.f,0.f,0.f,0.f,0.f,0.f};
#pragma unroll
    for (int nt = 0; nt < 2; ++nt) {
        v16h bf = *(const v16h*)(w1p + (size_t)(nt * 16 + n) * 32 + 16 * half);
        acc[nt] = __builtin_amdgcn_wmma_f32_16x16x32_f16(
            false, a.v, false, bf, (short)0, acc[nt], false, false);
    }

    float part[8];
#pragma unroll
    for (int r = 0; r < 8; ++r) part[r] = 0.f;
#pragma unroll
    for (int nt = 0; nt < 2; ++nt) {
        const int co = nt * 16 + n;
        const float bb = b1[co];
        const float ww = w2[co];
#pragma unroll
        for (int r = 0; r < 8; ++r)
            part[r] += tanhf(acc[nt][r] + bb) * ww;
    }
    // reduce over the 16 lanes of each half (columns of the tile)
#pragma unroll
    for (int off = 1; off < 16; off <<= 1)
#pragma unroll
        for (int r = 0; r < 8; ++r)
            part[r] += __shfl_xor(part[r], off, 32);

    if (n == 0) {
        const float bb2 = b2[0];
#pragma unroll
        for (int r = 0; r < 8; ++r)
            s[(size_t)b * 2048 + lo0 + r + 8 * half] = part[r] + bb2;
    }
}

// ---------------------------------------------------------------------------
// Grouped GRU scan.  One wave per batch element; lane o owns output channel o
// (group g=o>>3 reads the contiguous 12-float slice [12g,12g+12) of [xt,h]).
// ---------------------------------------------------------------------------
__global__ __launch_bounds__(256) void gru_scan(
    const f16* __restrict__ p1h,                   // (B, 2048, 16)
    const float* __restrict__ wz, const float* __restrict__ bz,
    const float* __restrict__ wr, const float* __restrict__ br,
    const float* __restrict__ wh, const float* __restrict__ bh,
    f16* __restrict__ gh)                          // (B, 2048, 32)
{
    __shared__ float lds[8][64];
    const int lane = threadIdx.x & 31;
    const int wave = threadIdx.x >> 5;
    const int b = blockIdx.x * 8 + wave;

    float Wz[12], Wr[12], Wh[12];
#pragma unroll
    for (int j = 0; j < 12; ++j) {
        Wz[j] = wz[(lane * 12 + j) * 5 + 2];       // center tap of k=5 conv
        Wr[j] = wr[(lane * 12 + j) * 5 + 2];
        Wh[j] = wh[(lane * 12 + j) * 5 + 2];
    }
    const float bzv = bz[lane], brv = br[lane], bhv = bh[lane];
    const int grp = lane >> 3;

    const f16* xrow = p1h + (size_t)b * 2048 * 16;
    f16* gout = gh + (size_t)b * 2048 * 32;

    float h = 0.f;
    for (int t = 0; t < 2048; ++t) {
        __syncthreads();
        if (lane < 16) lds[wave][lane] = (float)xrow[t * 16 + lane];
        lds[wave][16 + lane] = h;
        __syncthreads();

        float c[12];
#pragma unroll
        for (int j = 0; j < 12; ++j) c[j] = lds[wave][12 * grp + j];

        float az = bzv, ar = brv;
#pragma unroll
        for (int j = 0; j < 12; ++j) { az += Wz[j] * c[j]; ar += Wr[j] * c[j]; }
        const float z = 1.f / (1.f + expf(-az));
        const float r = 1.f / (1.f + expf(-ar));

        __syncthreads();
        lds[wave][16 + lane] = r * h;
        __syncthreads();
#pragma unroll
        for (int j = 0; j < 12; ++j) c[j] = lds[wave][12 * grp + j];

        float ah = bhv;
#pragma unroll
        for (int j = 0; j < 12; ++j) ah += Wh[j] * c[j];
        const float hh = tanhf(ah);

        h = (1.f - z) * h + z * hh;
        gout[t * 32 + lane] = (f16)h;
    }
}

// ---------------------------------------------------------------------------
// Softmax over time + attention context + classifier head.  One block/batch.
// ---------------------------------------------------------------------------
__global__ __launch_bounds__(256) void pool_cls(
    const float* __restrict__ s, const f16* __restrict__ gh,
    const float* __restrict__ cw1, const float* __restrict__ cb1,
    const float* __restrict__ cw2, const float* __restrict__ cb2,
    float* __restrict__ out)
{
    __shared__ float red[256];
    __shared__ float cbuf[256][32];
    __shared__ float ctxs[32];
    __shared__ float h1s[16];

    const int b = blockIdx.x, tid = threadIdx.x;
    const float* srow = s + (size_t)b * 2048;
    const f16* gb = gh + (size_t)b * 2048 * 32;

    float mx = -3.0e38f;
    for (int t = tid; t < 2048; t += 256) mx = fmaxf(mx, srow[t]);
    red[tid] = mx; __syncthreads();
    for (int o = 128; o > 0; o >>= 1) {
        if (tid < o) red[tid] = fmaxf(red[tid], red[tid + o]);
        __syncthreads();
    }
    mx = red[0]; __syncthreads();

    float se = 0.f;
    float ctx[32];
#pragma unroll
    for (int c = 0; c < 32; ++c) ctx[c] = 0.f;
    for (int t = tid; t < 2048; t += 256) {
        const float e = expf(srow[t] - mx);
        se += e;
#pragma unroll
        for (int c = 0; c < 32; ++c) ctx[c] += e * (float)gb[(size_t)t * 32 + c];
    }
    red[tid] = se; __syncthreads();
    for (int o = 128; o > 0; o >>= 1) {
        if (tid < o) red[tid] += red[tid + o];
        __syncthreads();
    }
    se = red[0];

#pragma unroll
    for (int c = 0; c < 32; ++c) cbuf[tid][c] = ctx[c];
    __syncthreads();
    for (int o = 128; o > 0; o >>= 1) {
        if (tid < o) {
#pragma unroll
            for (int c = 0; c < 32; ++c) cbuf[tid][c] += cbuf[tid + o][c];
        }
        __syncthreads();
    }
    if (tid < 32) ctxs[tid] = cbuf[0][tid] / se;
    __syncthreads();
    if (tid < 16) {
        float a = cb1[tid];
        for (int c = 0; c < 32; ++c) a += ctxs[c] * cw1[tid * 32 + c];
        h1s[tid] = gelu_exact(a);
    }
    __syncthreads();
    if (tid < 5) {
        float a = cb2[tid];
        for (int j = 0; j < 16; ++j) a += h1s[j] * cw2[tid * 16 + j];
        out[(size_t)b * 5 + tid] = a;
    }
}

// ---------------------------------------------------------------------------
// Elementwise helpers
// ---------------------------------------------------------------------------
__global__ __launch_bounds__(256) void pack_x(const float* __restrict__ x,
                                              f16* __restrict__ xh) {
    const int idx = blockIdx.x * 256 + threadIdx.x;     // (b, c, l), l fastest
    if (idx >= BATCH * 16 * 2048) return;
    const int l = idx & 2047;
    const int c = (idx >> 11) & 15;
    const int b = idx >> 15;
    xh[((size_t)b * 2048 + l) * 16 + c] = (f16)x[idx];
}

template <int C, int LIN>
__global__ __launch_bounds__(256) void dwconv(const f16* __restrict__ in,
                                              const float* __restrict__ dww,
                                              const float* __restrict__ dwb,
                                              f16* __restrict__ out) {
    const int idx = blockIdx.x * 256 + threadIdx.x;
    if (idx >= BATCH * LIN * C) return;
    const int c = idx % C;
    const int l = (idx / C) % LIN;
    const int b = idx / (C * LIN);
    const f16* ib = in + (size_t)b * LIN * C;
    float acc = dwb[c];
#pragma unroll
    for (int dt = 0; dt < 3; ++dt) {
        const int li = l + dt - 1;
        if (li >= 0 && li < LIN) acc += (float)ib[(size_t)li * C + c] * dww[c * 3 + dt];
    }
    out[idx] = (f16)acc;
}

// out[b][j][c] = lerp(y, pos=j*(LIN-1)/(2LIN-1)) + lateral
template <int C, int LIN, bool LAT_F32_NCL>
__global__ __launch_bounds__(256) void upsample_add(const f16* __restrict__ y,
                                                    const void* __restrict__ lat,
                                                    f16* __restrict__ out) {
    constexpr int LOUT = 2 * LIN;
    const int idx = blockIdx.x * 256 + threadIdx.x;
    if (idx >= BATCH * LOUT * C) return;
    const int c = idx % C;
    const int j = (idx / C) % LOUT;
    const int b = idx / (C * LOUT);
    const float pos = (float)j * ((float)(LIN - 1) / (float)(LOUT - 1));
    int i0 = (int)pos;
    int i1 = i0 + 1 < LIN ? i0 + 1 : LIN - 1;
    const float f = pos - (float)i0;
    const f16* yb = y + (size_t)b * LIN * C;
    float v = (float)yb[(size_t)i0 * C + c] * (1.f - f) + (float)yb[(size_t)i1 * C + c] * f;
    float lv;
    if (LAT_F32_NCL)
        lv = ((const float*)lat)[((size_t)b * C + c) * LOUT + j];
    else
        lv = (float)((const f16*)lat)[((size_t)b * LOUT + j) * C + c];
    out[idx] = (f16)(v + lv);
}

// ---------------------------------------------------------------------------
// Parameter packing
// ---------------------------------------------------------------------------
template <int CIN, int COUT, int KSZ, int KPAD>
__global__ __launch_bounds__(256) void pack_conv_w(const float* __restrict__ w,
                                                   f16* __restrict__ wp) {
    const int idx = blockIdx.x * 256 + threadIdx.x;
    if (idx >= COUT * KPAD) return;
    const int co = idx / KPAD, k = idx % KPAD;
    float v = 0.f;
    if (k < KSZ * CIN) {
        const int t = k / CIN, ci = k % CIN;
        v = w[(co * CIN + ci) * KSZ + t];
    }
    wp[idx] = (f16)v;
}

__global__ __launch_bounds__(256) void cvt_f32_to_f16(const float* __restrict__ src,
                                                      f16* __restrict__ dst, int n) {
    const int i = blockIdx.x * 256 + threadIdx.x;
    if (i < n) dst[i] = (f16)src[i];
}

// y = acc*A + B  with  A = g/sqrt(v+eps),  B = bnb - m*A + convb*A
__global__ void fuse_bn(const float* __restrict__ g, const float* __restrict__ bb,
                        const float* __restrict__ m, const float* __restrict__ v,
                        const float* __restrict__ cb, float* __restrict__ A,
                        float* __restrict__ B, int C) {
    const int i = threadIdx.x;
    if (i < C) {
        const float a = g[i] * rsqrtf(v[i] + 1e-5f);
        A[i] = a;
        B[i] = bb[i] - m[i] * a + cb[i] * a;
    }
}

// ---------------------------------------------------------------------------
extern "C" void kernel_launch(void* const* d_in, const int* in_sizes, int n_in,
                              void* d_out, int out_size, void* d_ws, size_t ws_size,
                              hipStream_t stream) {
    (void)in_sizes; (void)n_in; (void)out_size; (void)ws_size;
    // --- input pointers (setup_inputs flattened order) ---
    const float* x    = (const float*)d_in[0];
    const float* e1w  = (const float*)d_in[1];  const float* e1b = (const float*)d_in[2];
    const float* e2w  = (const float*)d_in[3];  const float* e2b = (const float*)d_in[4];
    const float* e3w  = (const float*)d_in[5];  const float* e3b = (const float*)d_in[6];
    const float* bn1g = (const float*)d_in[7];  const float* bn1b = (const float*)d_in[8];
    const float* bn1m = (const float*)d_in[9];  const float* bn1v = (const float*)d_in[10];
    const float* bn2g = (const float*)d_in[11]; const float* bn2b = (const float*)d_in[12];
    const float* bn2m = (const float*)d_in[13]; const float* bn2v = (const float*)d_in[14];
    const float* bn3g = (const float*)d_in[15]; const float* bn3b = (const float*)d_in[16];
    const float* bn3m = (const float*)d_in[17]; const float* bn3v = (const float*)d_in[18];
    const float* f3dw = (const float*)d_in[19]; const float* f3dwb = (const float*)d_in[20];
    const float* f3pw = (const float*)d_in[21]; const float* f3pwb = (const float*)d_in[22];
    const float* f3g  = (const float*)d_in[23]; const float* f3b = (const float*)d_in[24];
    const float* f3m  = (const float*)d_in[25]; const float* f3v = (const float*)d_in[26];
    const float* f2dw = (const float*)d_in[27]; const float* f2dwb = (const float*)d_in[28];
    const float* f2pw = (const float*)d_in[29]; const float* f2pwb = (const float*)d_in[30];
    const float* f2g  = (const float*)d_in[31]; const float* f2b = (const float*)d_in[32];
    const float* f2m  = (const float*)d_in[33]; const float* f2v = (const float*)d_in[34];
    const float* f1dw = (const float*)d_in[35]; const float* f1dwb = (const float*)d_in[36];
    const float* f1pw = (const float*)d_in[37]; const float* f1pwb = (const float*)d_in[38];
    const float* f1g  = (const float*)d_in[39]; const float* f1b = (const float*)d_in[40];
    const float* f1m  = (const float*)d_in[41]; const float* f1v = (const float*)d_in[42];
    const float* gwz  = (const float*)d_in[43]; const float* gbz = (const float*)d_in[44];
    const float* gwr  = (const float*)d_in[45]; const float* gbr = (const float*)d_in[46];
    const float* gwh  = (const float*)d_in[47]; const float* gbh = (const float*)d_in[48];
    const float* aw1  = (const float*)d_in[49]; const float* ab1 = (const float*)d_in[50];
    const float* aw2  = (const float*)d_in[51]; const float* ab2 = (const float*)d_in[52];
    const float* cw1  = (const float*)d_in[53]; const float* cb1 = (const float*)d_in[54];
    const float* cw2  = (const float*)d_in[55]; const float* cb2 = (const float*)d_in[56];
    float* out = (float*)d_out;

    // --- workspace layout (bytes), with dataflow-safe reuse ---
    char* ws = (char*)d_ws;
    constexpr size_t MiB = 1024ull * 1024ull;
    f16* xh   = (f16*)(ws + 0);         // (B,2048,16)  8 MiB   [dead after enc1]
    f16* dw3o = (f16*)(ws + 0);         // (B,256,128)  8 MiB   reuse of xh
    f16* c1h  = (f16*)(ws + 8 * MiB);   // (B,1024,32)  8 MiB
    f16* c2h  = (f16*)(ws + 16 * MiB);  // (B,512,64)   8 MiB
    f16* c3h  = (f16*)(ws + 24 * MiB);  // (B,256,128)  8 MiB   [dead after dw3]
    f16* dw2o = (f16*)(ws + 24 * MiB);  // (B,512,64)   reuse of c3h
    f16* y3h  = (f16*)(ws + 32 * MiB);  // (B,256,64)   4 MiB   [dead after up3]
    f16* y2h  = (f16*)(ws + 32 * MiB);  // (B,512,32)   reuse of y3h
    f16* p3h  = (f16*)(ws + 36 * MiB);  // (B,512,64)   8 MiB   [dead after dw2]
    f16* dw1o = (f16*)(ws + 36 * MiB);  // (B,1024,32)  reuse of p3h
    f16* p2h  = (f16*)(ws + 44 * MiB);  // (B,1024,32)  8 MiB   [dead after dw1]
    f16* y1h  = (f16*)(ws + 44 * MiB);  // (B,1024,16)  reuse of p2h
    f16* p1h  = (f16*)(ws + 52 * MiB);  // (B,2048,16)  8 MiB
    f16* gh   = (f16*)(ws + 60 * MiB);  // (B,2048,32) 16 MiB
    float* sb = (float*)(ws + 76 * MiB);// (B,2048)     1 MiB
    char* wb  = ws + 77 * MiB;          // packed weights ~120 KiB
    f16* enc1p = (f16*)(wb);            // 32*256
    f16* enc2p = (f16*)(wb + 16384);    // 64*224
    f16* enc3p = (f16*)(wb + 45056);    // 128*192
    f16* pw3p  = (f16*)(wb + 94208);    // 64*128
    f16* pw2p  = (f16*)(wb + 110592);   // 32*64
    f16* pw1p  = (f16*)(wb + 114688);   // 16*32
    f16* aw1p  = (f16*)(wb + 115712);   // 32*32
    float* bnq = (float*)(wb + 117760);
    float* A1 = bnq;        float* B1 = bnq + 32;
    float* A2 = bnq + 64;   float* B2 = bnq + 128;
    float* A3 = bnq + 192;  float* B3 = bnq + 320;
    float* AF3 = bnq + 448; float* BF3 = bnq + 512;
    float* AF2 = bnq + 576; float* BF2 = bnq + 608;
    float* AF1 = bnq + 640; float* BF1 = bnq + 656;

    // --- parameter packing / BN fusion (tiny) ---
    pack_conv_w<16, 32, 15, 256><<<(32 * 256 + 255) / 256, 256, 0, stream>>>(e1w, enc1p);
    pack_conv_w<32, 64, 7, 224><<<(64 * 224 + 255) / 256, 256, 0, stream>>>(e2w, enc2p);
    pack_conv_w<64, 128, 3, 192><<<(128 * 192 + 255) / 256, 256, 0, stream>>>(e3w, enc3p);
    cvt_f32_to_f16<<<(8192 + 255) / 256, 256, 0, stream>>>(f3pw, pw3p, 8192);
    cvt_f32_to_f16<<<(2048 + 255) / 256, 256, 0, stream>>>(f2pw, pw2p, 2048);
    cvt_f32_to_f16<<<(512 + 255) / 256, 256, 0, stream>>>(f1pw, pw1p, 512);
    cvt_f32_to_f16<<<(1024 + 255) / 256, 256, 0, stream>>>(aw1, aw1p, 1024);
    fuse_bn<<<1, 128, 0, stream>>>(bn1g, bn1b, bn1m, bn1v, e1b, A1, B1, 32);
    fuse_bn<<<1, 128, 0, stream>>>(bn2g, bn2b, bn2m, bn2v, e2b, A2, B2, 64);
    fuse_bn<<<1, 128, 0, stream>>>(bn3g, bn3b, bn3m, bn3v, e3b, A3, B3, 128);
    fuse_bn<<<1, 128, 0, stream>>>(f3g, f3b, f3m, f3v, f3pwb, AF3, BF3, 64);
    fuse_bn<<<1, 128, 0, stream>>>(f2g, f2b, f2m, f2v, f2pwb, AF2, BF2, 32);
    fuse_bn<<<1, 128, 0, stream>>>(f1g, f1b, f1m, f1v, f1pwb, AF1, BF1, 16);

    // --- input transpose to channel-last f16 ---
    pack_x<<<16384, 256, 0, stream>>>(x, xh);

    // --- encoder (WMMA implicit GEMM, fused bias+BN+GELU) ---
    conv_wmma<16, 32, 15, 2, 7, 2048, 1024, 8><<<1024, 256, 0, stream>>>(xh, enc1p, A1, B1, c1h);
    conv_wmma<32, 64, 7, 2, 3, 1024, 512, 7><<<512, 256, 0, stream>>>(c1h, enc2p, A2, B2, c2h);
    conv_wmma<64, 128, 3, 2, 1, 512, 256, 6><<<256, 256, 0, stream>>>(c2h, enc3p, A3, B3, c3h);

    // --- FPN level 3: dw -> pw(WMMA) -> upsample+lateral(c2) ---
    dwconv<128, 256><<<16384, 256, 0, stream>>>(c3h, f3dw, f3dwb, dw3o);
    conv_wmma<128, 64, 1, 1, 0, 256, 256, 4><<<256, 256, 0, stream>>>(dw3o, pw3p, AF3, BF3, y3h);
    upsample_add<64, 256, false><<<16384, 256, 0, stream>>>(y3h, c2h, p3h);

    // --- FPN level 2 ---
    dwconv<64, 512><<<16384, 256, 0, stream>>>(p3h, f2dw, f2dwb, dw2o);
    conv_wmma<64, 32, 1, 1, 0, 512, 512, 2><<<512, 256, 0, stream>>>(dw2o, pw2p, AF2, BF2, y2h);
    upsample_add<32, 512, false><<<16384, 256, 0, stream>>>(y2h, c1h, p2h);

    // --- FPN level 1 (lateral is the raw f32 NCL input) ---
    dwconv<32, 1024><<<16384, 256, 0, stream>>>(p2h, f1dw, f1dwb, dw1o);
    conv_wmma<32, 16, 1, 1, 0, 1024, 1024, 1><<<1024, 256, 0, stream>>>(dw1o, pw1p, AF1, BF1, y1h);
    upsample_add<16, 1024, true><<<16384, 256, 0, stream>>>(y1h, (const void*)x, p1h);

    // --- GRU scan (one wave per batch element) ---
    gru_scan<<<16, 256, 0, stream>>>(p1h, gwz, gbz, gwr, gbr, gwh, gbh, gh);

    // --- attention scores (WMMA) + softmax/context/classifier ---
    attn_score<<<2048, 256, 0, stream>>>(gh, aw1p, ab1, aw2, ab2, sb);
    pool_cls<<<BATCH, 256, 0, stream>>>(sb, gh, cw1, cb1, cw2, cb2, out);
}
